// HyperModulatedLinear_42167988912486
// MI455X (gfx1250) — matrix operand order repaired and checked by reading, not compiled, not run on hardware
//
#include <hip/hip_runtime.h>

#define BATCH   128
#define IN_DIM  1024
#define OUT_DIM 1024
#define ZD_DIM  128
#define KTOT    (IN_DIM * ZD_DIM)   // 131072 = GEMM K

typedef __attribute__((ext_vector_type(16))) __bf16 v16bf;
typedef __attribute__((ext_vector_type(8)))  float  v8f;

union ABFrag { unsigned u[8]; v16bf v; };

// fp32 pair -> packed bf16 (RNE), single VALU op on gfx1250
__device__ __forceinline__ unsigned pk_bf16(float a, float b) {
    unsigned r;
    asm("v_cvt_pk_bf16_f32 %0, %1, %2" : "=v"(r) : "v"(a), "v"(b));
    return r;
}
// packed bf16 multiply (A fragment = x[b,i] * z[b,k..k+1])
__device__ __forceinline__ unsigned pk_mul_bf16(unsigned a, unsigned b) {
    unsigned r;
    asm("v_pk_mul_bf16 %0, %1, %2" : "=v"(r) : "v"(a), "v"(b));
    return r;
}

__global__ __launch_bounds__(256)
void hyper_linear_kernel(const float* __restrict__ xg,
                         const float* __restrict__ zg,
                         const float* __restrict__ bw,
                         const float* __restrict__ dwW,
                         const float* __restrict__ dwb,
                         const float* __restrict__ bbias,
                         const float* __restrict__ dbW,
                         const float* __restrict__ dbb,
                         float* __restrict__ out) {
    // LDS: bf16(x) rows (16 x 1024, u16), bf16(z) rows as pairs (16 x 64 u32),
    // plus per-wave partial-tile reduction buffer.  44 KB total.
    __shared__ __align__(16) unsigned short xpk[16 * IN_DIM];      // 32 KB
    __shared__ __align__(16) unsigned       zpk[16 * (ZD_DIM/2)];  //  4 KB
    __shared__ float red[8][16][16];                               //  8 KB

    const int t  = threadIdx.x;
    const int bx = blockIdx.x;
    const int rt = bx & 7;                 // batch tile (low bits -> L2 reuse of Wf stripe)
    const int ct = bx >> 3;                // out tile
    const int b0 = rt * 16;
    const int o0 = ct * 16;

    // ---- stage bf16(x), bf16(z) into LDS (round once, reuse everywhere) ----
    {
        unsigned* xp32 = (unsigned*)xpk;   // write as bf16 pairs
        for (int q = 0; q < 32; ++q) {     // 16*512 = 8192 pairs
            int pi = t + q * 256;
            int m  = pi >> 9;
            int ip = pi & 511;
            float2 f = *(const float2*)(xg + (size_t)(b0 + m) * IN_DIM + 2 * ip);
            xp32[pi] = pk_bf16(f.x, f.y);
        }
        for (int q = 0; q < 4; ++q) {      // 16*64 = 1024 pairs
            int pi = t + q * 256;
            int m  = pi >> 6;
            int ip = pi & 63;
            float2 f = *(const float2*)(zg + (size_t)(b0 + m) * ZD_DIM + 2 * ip);
            zpk[pi] = pk_bf16(f.x, f.y);
        }
    }
    __syncthreads();

    // ---- streamed GEMM: D += (x ⊗ z) @ Wf^T, bf16 WMMA, fp32 accum ----
    const int wave = t >> 5;
    const int lane = t & 31;
    const int hh   = lane >> 4;            // lane half
    const int mn   = lane & 15;            // A row (m) / B col (n)

    const int jbase = wave * (KTOT / 8);   // per-wave K chunk of 16384
    const int ibase = jbase >> 7;          // starting i (K = i*128 + k)

    // B fragment: lanes0-15 hold K 0..15, lanes16-31 hold K 16..31 (pairs per VGPR)
    const float* bp0 = dwW + (size_t)(o0 + mn) * KTOT + (size_t)jbase + hh * 16;

    v8f acc = {};
    for (int ii = 0; ii < 128; ++ii) {
        unsigned xs = xpk[mn * IN_DIM + ibase + ii];   // bf16 bits of x[b0+mn, i]
        xs |= xs << 16;                                // replicate into both halves
        const float* bpi = bp0 + ii * 128;
        __builtin_prefetch(bpi + 8 * 128, 0, 0);       // global_prefetch of Wf stream
#pragma unroll
        for (int p = 0; p < 4; ++p) {                  // kk0 = 32*p within z row
            const float* bp = bpi + 32 * p;
            float4 f0 = *(const float4*)(bp);
            float4 f1 = *(const float4*)(bp + 4);
            float4 f2 = *(const float4*)(bp + 8);
            float4 f3 = *(const float4*)(bp + 12);
            ABFrag bb;
            bb.u[0] = pk_bf16(f0.x, f0.y); bb.u[1] = pk_bf16(f0.z, f0.w);
            bb.u[2] = pk_bf16(f1.x, f1.y); bb.u[3] = pk_bf16(f1.z, f1.w);
            bb.u[4] = pk_bf16(f2.x, f2.y); bb.u[5] = pk_bf16(f2.z, f2.w);
            bb.u[6] = pk_bf16(f3.x, f3.y); bb.u[7] = pk_bf16(f3.z, f3.w);

            // A fragment (contiguous k order, same as B): half h -> K {16h..16h+7} (v0-3), {16h+8..16h+15} (v4-7)
            uint4 za = *(const uint4*)&zpk[mn * 64 + 16 * p + 8 * hh];
            uint4 zb = *(const uint4*)&zpk[mn * 64 + 16 * p + 8 * hh + 4];
            ABFrag aa;
            aa.u[0] = pk_mul_bf16(xs, za.x); aa.u[1] = pk_mul_bf16(xs, za.y);
            aa.u[2] = pk_mul_bf16(xs, za.z); aa.u[3] = pk_mul_bf16(xs, za.w);
            aa.u[4] = pk_mul_bf16(xs, zb.x); aa.u[5] = pk_mul_bf16(xs, zb.y);
            aa.u[6] = pk_mul_bf16(xs, zb.z); aa.u[7] = pk_mul_bf16(xs, zb.w);

            acc = __builtin_amdgcn_wmma_f32_16x16x32_bf16(
                      false, aa.v, false, bb.v, (short)0, acc, false, false);
        }
    }

    // C/D layout: lane half h holds rows 8h..8h+7 (vgpr r -> row 8h+r), col = mn
#pragma unroll
    for (int r = 0; r < 8; ++r) red[wave][hh * 8 + r][mn] = acc[r];
    __syncthreads();

    // ---- epilogue: reduce K-chunks + base/bias terms (fp32, L2-resident) ----
    const int em = t >> 4;
    const int en = t & 15;
    float s = bbias[o0 + en] + dbb[o0 + en];
#pragma unroll
    for (int w = 0; w < 8; ++w) s += red[w][em][en];

    const float4* bw4  = (const float4*)(bw  + (size_t)(o0 + en) * IN_DIM);
    const float4* dwb4 = (const float4*)(dwb + (size_t)(o0 + en) * IN_DIM);
    const float4* xr4  = (const float4*)(xg  + (size_t)(b0 + em) * IN_DIM);
    float sb = 0.f;
    for (int k = 0; k < IN_DIM / 4; ++k) {
        float4 a = bw4[k], c = dwb4[k], xx = xr4[k];
        sb += (a.x + c.x) * xx.x + (a.y + c.y) * xx.y +
              (a.z + c.z) * xx.z + (a.w + c.w) * xx.w;
    }
    const float4* db4 = (const float4*)(dbW + (size_t)(o0 + en) * ZD_DIM);
    const float4* zr4 = (const float4*)(zg  + (size_t)(b0 + em) * ZD_DIM);
    for (int k = 0; k < ZD_DIM / 4; ++k) {
        float4 a = db4[k], zz = zr4[k];
        sb += a.x * zz.x + a.y * zz.y + a.z * zz.z + a.w * zz.w;
    }
    out[(size_t)(b0 + em) * OUT_DIM + (o0 + en)] = s + sb;
}

extern "C" void kernel_launch(void* const* d_in, const int* in_sizes, int n_in,
                              void* d_out, int out_size, void* d_ws, size_t ws_size,
                              hipStream_t stream) {
    const float* x     = (const float*)d_in[0];
    const float* z     = (const float*)d_in[1];
    const float* bw    = (const float*)d_in[2];
    const float* dwW   = (const float*)d_in[3];
    const float* dwb   = (const float*)d_in[4];
    const float* bbias = (const float*)d_in[5];
    const float* dbW   = (const float*)d_in[6];
    const float* dbb   = (const float*)d_in[7];
    float* out = (float*)d_out;

    // 8 batch-tiles x 64 out-tiles; batch tile in low bits of blockIdx
    hyper_linear_kernel<<<512, 256, 0, stream>>>(x, z, bw, dwW, dwb, bbias, dbW, dbb, out);
}
